// Decoder_53455162966720
// MI455X (gfx1250) — matrix operand order repaired and checked by reading, not compiled, not run on hardware
//
#include <hip/hip_runtime.h>
#include <hip/hip_bf16.h>
#include <math.h>

typedef __bf16 bf16;
typedef __attribute__((ext_vector_type(16))) __bf16 v16bf;
typedef __attribute__((ext_vector_type(8)))  float  v8f;

#define Vv 50000
#define Ee 300
#define Hh 600
#define Bb 32
#define Ss 400
#define Tt 24
#define KX 1504   // (E + H + H)=1500 padded to multiple of 32
#define KH 608    // H=600 padded
#define KC 1216   // 2H=1200 padded
#define NG 2400   // 4H

union FragU { uint4 u[2]; v16bf v; };
union F16U  { float4 f4[4]; float f[16]; };

// ---------- WMMA fragment primitives (wave32; same contiguous k order for A and B) ----------
// p must already be the lane-resolved base; tile/chunk selection via constant offsets.
// A 16-bit 16x32: 16 consecutive k per lane (elements 0..7 at p, 8..15 at p+8)
__device__ __forceinline__ v16bf load_a_frag(const bf16* p) {
  FragU u;
  u.u[0] = *(const uint4*)(p);
  u.u[1] = *(const uint4*)(p + 8);
  return u.v;
}
// B 16-bit 32x16: 16 consecutive k per lane
__device__ __forceinline__ v16bf load_b_frag(const bf16* p) {
  FragU u;
  u.u[0] = *(const uint4*)(p);
  u.u[1] = *(const uint4*)(p + 8);
  return u.v;
}

// lane base helpers
__device__ __forceinline__ const bf16* a_base(const bf16* A, int Ks) {
  int lane = threadIdx.x & 31;
  return A + (lane & 15) * Ks + (lane >> 4) * 16;
}
__device__ __forceinline__ const bf16* b_base(const bf16* W, int n0, int Ks) {
  int lane = threadIdx.x & 31;
  return W + (long)(n0 + (lane & 15)) * Ks + (lane >> 4) * 16;
}

// fallback: fp32 W_gen converted on the fly (K stride = 600, unpadded)
__device__ __forceinline__ v16bf load_b_f32(const float* W, int n0, int kc) {
  int lane = threadIdx.x & 31;
  int half = lane >> 4;
  long col = n0 + (lane & 15);
  long idx = col * (long)Hh + kc + half * 16;
  long maxidx = (long)Vv * (long)Hh - 16;
  idx = idx < maxidx ? idx : maxidx;   // tail clamp; A is zero-padded there anyway
  const float* p = W + idx;
  F16U t;
  t.f4[0] = *(const float4*)(p);
  t.f4[1] = *(const float4*)(p + 4);
  t.f4[2] = *(const float4*)(p + 8);
  t.f4[3] = *(const float4*)(p + 12);
  v16bf r;
#pragma unroll
  for (int e = 0; e < 16; ++e) r[e] = (bf16)t.f[e];
  return r;
}

__device__ __forceinline__ v8f wmma_bf16(v16bf a, v16bf b, v8f c) {
  return __builtin_amdgcn_wmma_f32_16x16x32_bf16(false, a, false, b, (short)0, c, false, false);
}

// ---------------------------- prep kernels (once per launch) ----------------------------
__global__ void k_prep_wcat(const float* Wih, const float* Whh, const float* bih,
                            const float* bhh, bf16* wcat, float* biasc) {
  int idx = blockIdx.x * blockDim.x + threadIdx.x;
  if (idx < NG) biasc[idx] = bih[idx] + bhh[idx];
  if (idx >= NG * KX) return;
  int j = idx / KX, k = idx % KX;
  float v = 0.f;
  if (k < Ee + Hh)           v = Wih[j * (Ee + Hh) + k];
  else if (k < Ee + 2 * Hh)  v = Whh[j * Hh + (k - (Ee + Hh))];
  wcat[idx] = (bf16)v;
}

__global__ void k_prep_wai(const float* Wai, bf16* wai) {
  int idx = blockIdx.x * blockDim.x + threadIdx.x;
  if (idx >= KH * KH) return;
  int r = idx / KH, k = idx % KH;
  wai[idx] = (bf16)((r < Hh && k < Hh) ? Wai[r * Hh + k] : 0.f);
}

__global__ void k_prep_wao(const float* Wao, bf16* wao) {
  int idx = blockIdx.x * blockDim.x + threadIdx.x;
  if (idx >= KH * KC) return;
  int r = idx / KC, k = idx % KC;
  wao[idx] = (bf16)((r < Hh && k < 2 * Hh) ? Wao[r * 2 * Hh + k] : 0.f);
}

// W_gen fp32 [50000,600] -> zero-padded bf16 [50000,608] (only when ws fits)
__global__ void k_prep_wgen(const float* Wgen, bf16* wgenb) {
  long idx = (long)blockIdx.x * blockDim.x + threadIdx.x;
  if (idx >= (long)Vv * KH) return;
  long r = idx / KH;
  int  k = (int)(idx % KH);
  wgenb[idx] = (bf16)((k < Hh) ? Wgen[r * Hh + k] : 0.f);
}

__global__ void k_init_state(const float* ench, const float* encc,
                             float* h, float* c, float* feed) {
  int idx = blockIdx.x * blockDim.x + threadIdx.x;
  if (idx >= Bb * Hh) return;
  int b = idx / Hh, k = idx % Hh;
  int half = Hh / 2;
  int src = (k < half) ? (b * half + k) : (Bb * half + b * half + (k - half));
  h[idx] = ench[src];
  c[idx] = encc[src];
  feed[idx] = 0.f;
}

// ---------------------------- per-step kernels ----------------------------
__global__ void k_build_xcat(const int* ques, const float* emb, const float* feed,
                             const float* h, bf16* xcat, int t) {
  int idx = blockIdx.x * blockDim.x + threadIdx.x;
  if (idx >= Bb * KX) return;
  int b = idx / KX, k = idx % KX;
  float v = 0.f;
  if (k < Ee) {
    int q = ques[b * Tt + t];
    v = emb[(long)q * Ee + k];
  } else if (k < Ee + Hh) {
    v = feed[b * Hh + (k - Ee)];
  } else if (k < Ee + 2 * Hh) {
    v = h[b * Hh + (k - Ee - Hh)];
  }
  xcat[idx] = (bf16)v;
}

// gates[32,2400] = xcat[32,1504] @ wcat[2400,1504]^T + biasc
// grid 30, block 32: one wave = 2 M-tiles x 5 N-tiles; all B frags of a chunk
// staged into registers first so one load clause covers the whole chunk.
__global__ void __launch_bounds__(32, 1)
k_gemm_gates(const bf16* xcat, const bf16* wcat, const float* biasc, float* gates) {
  int n0 = blockIdx.x * 80;
  const bf16* ab = a_base(xcat, KX);
  const bf16* bb = b_base(wcat, n0, KX);
  v8f acc[2][5] = {};
  for (int kc = 0; kc < KX; kc += 32) {
    v16bf a0 = load_a_frag(ab + kc);
    v16bf a1 = load_a_frag(ab + 16 * KX + kc);
    v16bf bfr[5];
#pragma unroll
    for (int i = 0; i < 5; ++i) bfr[i] = load_b_frag(bb + i * 16 * KX + kc);
#pragma unroll
    for (int i = 0; i < 5; ++i) {
      acc[0][i] = wmma_bf16(a0, bfr[i], acc[0][i]);
      acc[1][i] = wmma_bf16(a1, bfr[i], acc[1][i]);
    }
  }
  int lane = threadIdx.x & 31;
#pragma unroll
  for (int i = 0; i < 5; ++i) {
    int n = n0 + i * 16 + (lane & 15);
    float bs = biasc[n];
#pragma unroll
    for (int mt = 0; mt < 2; ++mt)
#pragma unroll
      for (int r = 0; r < 8; ++r) {
        int m = mt * 16 + r + ((lane >> 4) << 3);
        gates[m * NG + n] = acc[mt][i][r] + bs;
      }
  }
}

__global__ void k_lstm_ew(const float* gates, float* h, float* c,
                          bf16* h_bf, bf16* ctxh) {
  int idx = blockIdx.x * blockDim.x + threadIdx.x;
  if (idx >= Bb * 624) return;
  int b = idx / 624, k = idx % 624;
  if (k < Hh) {
    const float* g = gates + b * NG;
    float i_ = g[k], f_ = g[Hh + k], g_ = g[2 * Hh + k], o_ = g[3 * Hh + k];
    float si = 1.f / (1.f + expf(-i_));
    float sf = 1.f / (1.f + expf(-f_));
    float so = 1.f / (1.f + expf(-o_));
    float cn = sf * c[b * Hh + k] + si * tanhf(g_);
    float hn = so * tanhf(cn);
    c[b * Hh + k] = cn;
    h[b * Hh + k] = hn;
    h_bf[b * KH + k] = (bf16)hn;
    ctxh[b * KC + Hh + k] = (bf16)hn;     // h half of [ctx | h]
  } else if (k < KH) {
    h_bf[b * KH + k] = (bf16)0.f;         // K padding
  } else {
    ctxh[b * KC + 2 * Hh + (k - KH)] = (bf16)0.f;  // K padding (1200..1215)
  }
}

// q[32,608] = h_bf[32,608] @ wai[608,608]^T  (grid 19, block 32, 2M x 2N per wave)
__global__ void __launch_bounds__(32, 1)
k_gemm_q(const bf16* h_bf, const bf16* wai, float* q) {
  int n0 = blockIdx.x * 32;
  const bf16* ab = a_base(h_bf, KH);
  const bf16* bb = b_base(wai, n0, KH);
  v8f acc[2][2] = {};
  for (int kc = 0; kc < KH; kc += 32) {
    v16bf a0 = load_a_frag(ab + kc);
    v16bf a1 = load_a_frag(ab + 16 * KH + kc);
    v16bf bfr[2];
#pragma unroll
    for (int i = 0; i < 2; ++i) bfr[i] = load_b_frag(bb + i * 16 * KH + kc);
#pragma unroll
    for (int i = 0; i < 2; ++i) {
      acc[0][i] = wmma_bf16(a0, bfr[i], acc[0][i]);
      acc[1][i] = wmma_bf16(a1, bfr[i], acc[1][i]);
    }
  }
  int lane = threadIdx.x & 31;
#pragma unroll
  for (int i = 0; i < 2; ++i) {
    int n = n0 + i * 16 + (lane & 15);
#pragma unroll
    for (int mt = 0; mt < 2; ++mt)
#pragma unroll
      for (int r = 0; r < 8; ++r) {
        int m = mt * 16 + r + ((lane >> 4) << 3);
        q[m * KH + n] = acc[mt][i][r];
      }
  }
}

// align[b,s] = dot(q[b,:600], enc[b,s,:])  — one wave per (b,s)
__global__ void k_attn_score(const float* q, const float* enc, float* align) {
  int wid = (blockIdx.x * blockDim.x + threadIdx.x) >> 5;
  int lane = threadIdx.x & 31;
  if (wid >= Bb * Ss) return;
  int b = wid / Ss, s = wid % Ss;
  const float* qp = q + b * KH;
  const float* ep = enc + ((long)b * Ss + s) * Hh;
  float acc = 0.f;
  for (int k = lane; k < Hh; k += 32) acc += qp[k] * ep[k];
#pragma unroll
  for (int m = 16; m >= 1; m >>= 1) acc += __shfl_xor(acc, m, 32);
  if (lane == 0) align[b * Ss + s] = acc;
}

// softmax over S then ctx[b,:] = sum_s a[s]*enc[b,s,:] — one block per b
__global__ void k_softmax_ctx(const float* align, const float* enc, bf16* ctxh) {
  int b = blockIdx.x, tid = threadIdx.x;
  __shared__ float a_sh[Ss];
  __shared__ float red[512];
  float lmax = -3.0e38f;
  for (int s = tid; s < Ss; s += 512) {
    float v = align[b * Ss + s];
    a_sh[s] = v;
    lmax = fmaxf(lmax, v);
  }
  red[tid] = lmax; __syncthreads();
  for (int off = 256; off; off >>= 1) {
    if (tid < off) red[tid] = fmaxf(red[tid], red[tid + off]);
    __syncthreads();
  }
  float mx = red[0]; __syncthreads();
  float lsum = 0.f;
  for (int s = tid; s < Ss; s += 512) {
    float e = expf(a_sh[s] - mx);
    a_sh[s] = e;
    lsum += e;
  }
  red[tid] = lsum; __syncthreads();
  for (int off = 256; off; off >>= 1) {
    if (tid < off) red[tid] += red[tid + off];
    __syncthreads();
  }
  float inv = 1.f / red[0]; __syncthreads();
  for (int s = tid; s < Ss; s += 512) a_sh[s] *= inv;
  __syncthreads();
  for (int d = tid; d < Hh; d += 512) {
    float acc = 0.f;
    const float* ep = enc + ((long)b * Ss) * Hh + d;
    for (int s = 0; s < Ss; ++s) acc += a_sh[s] * ep[(long)s * Hh];
    ctxh[b * KC + d] = (bf16)acc;
  }
}

// attn_h = tanh(ctxh[32,1216] @ wao[608,1216]^T); write feed fp32 + attnh bf16
__global__ void __launch_bounds__(32, 1)
k_gemm_attnout(const bf16* ctxh, const bf16* wao, float* feed, bf16* attnh) {
  int n0 = blockIdx.x * 32;
  const bf16* ab = a_base(ctxh, KC);
  const bf16* bb = b_base(wao, n0, KC);
  v8f acc[2][2] = {};
  for (int kc = 0; kc < KC; kc += 32) {
    v16bf a0 = load_a_frag(ab + kc);
    v16bf a1 = load_a_frag(ab + 16 * KC + kc);
    v16bf bfr[2];
#pragma unroll
    for (int i = 0; i < 2; ++i) bfr[i] = load_b_frag(bb + i * 16 * KC + kc);
#pragma unroll
    for (int i = 0; i < 2; ++i) {
      acc[0][i] = wmma_bf16(a0, bfr[i], acc[0][i]);
      acc[1][i] = wmma_bf16(a1, bfr[i], acc[1][i]);
    }
  }
  int lane = threadIdx.x & 31;
#pragma unroll
  for (int i = 0; i < 2; ++i) {
    int n = n0 + i * 16 + (lane & 15);
#pragma unroll
    for (int mt = 0; mt < 2; ++mt)
#pragma unroll
      for (int r = 0; r < 8; ++r) {
        int m = mt * 16 + r + ((lane >> 4) << 3);
        float th = tanhf(acc[mt][i][r]);     // zero weight rows -> tanh(0)=0 padding
        attnh[m * KH + n] = (bf16)th;
        if (n < Hh) feed[m * Hh + n] = th;
      }
  }
}

// logits = attnh[32,608] @ wgenb[50000,608]^T + b_gen  (preconverted bf16 path)
// grid 625, block 32: one wave = 2 M-tiles x 5 N-tiles; W_gen streamed exactly once.
// Chunk loads staged as one clause (2 A + 5 B b128-pairs), then 10 WMMAs drain
// while the next chunk's clause is in flight.
__global__ void __launch_bounds__(32, 1)
k_gemm_logits_bf(const bf16* attnh, const bf16* wgenb, const float* bgen,
                 float* out, int t) {
  int n0 = blockIdx.x * 80;
  const bf16* ab = a_base(attnh, KH);
  const bf16* bb = b_base(wgenb, n0, KH);
  v8f acc[2][5] = {};
#pragma unroll
  for (int kc = 0; kc < KH; kc += 32) {
    v16bf a0 = load_a_frag(ab + kc);
    v16bf a1 = load_a_frag(ab + 16 * KH + kc);
    v16bf bfr[5];
#pragma unroll
    for (int i = 0; i < 5; ++i) bfr[i] = load_b_frag(bb + i * 16 * KH + kc);
#pragma unroll
    for (int i = 0; i < 5; ++i) {
      acc[0][i] = wmma_bf16(a0, bfr[i], acc[0][i]);
      acc[1][i] = wmma_bf16(a1, bfr[i], acc[1][i]);
    }
  }
  int lane = threadIdx.x & 31;
  float* orow = out + ((long)t * Bb) * (long)Vv;
#pragma unroll
  for (int i = 0; i < 5; ++i) {
    int n = n0 + i * 16 + (lane & 15);
    float bg = bgen[n];
#pragma unroll
    for (int mt = 0; mt < 2; ++mt)
#pragma unroll
      for (int r = 0; r < 8; ++r) {
        int m = mt * 16 + r + ((lane >> 4) << 3);
        orow[(long)m * Vv + n] = acc[mt][i][r] + bg;
      }
  }
}

// fallback: on-the-fly fp32->bf16 conversion of W_gen (grid (625,2))
__global__ void __launch_bounds__(32, 1)
k_gemm_logits_f32(const bf16* attnh, const float* Wgen, const float* bgen,
                  float* out, int t) {
  int n0 = blockIdx.x * 80;
  int mt = blockIdx.y;
  const bf16* ab = a_base(attnh, KH);
  v8f acc[5] = {};
  for (int kc = 0; kc < KH; kc += 32) {
    v16bf a = load_a_frag(ab + mt * 16 * KH + kc);
#pragma unroll
    for (int i = 0; i < 5; ++i) {
      v16bf b = load_b_f32(Wgen, n0 + i * 16, kc);
      acc[i] = wmma_bf16(a, b, acc[i]);
    }
  }
  int lane = threadIdx.x & 31;
  float* orow = out + ((long)t * Bb) * (long)Vv;
#pragma unroll
  for (int i = 0; i < 5; ++i) {
    int n = n0 + i * 16 + (lane & 15);
    float bg = bgen[n];
#pragma unroll
    for (int r = 0; r < 8; ++r) {
      int m = mt * 16 + r + ((lane >> 4) << 3);
      orow[(long)m * Vv + n] = acc[i][r] + bg;
    }
  }
}

// in-place log-softmax over V per (t,b) row — one block (1024 thr) per b
__global__ void k_logsoftmax(float* out, int t) {
  int b = blockIdx.x, tid = threadIdx.x;
  float* row = out + ((long)t * Bb + b) * (long)Vv;
  __shared__ float red[1024];
  float lmax = -3.0e38f;
  for (int i = tid; i < Vv; i += 1024) lmax = fmaxf(lmax, row[i]);
  red[tid] = lmax; __syncthreads();
  for (int off = 512; off; off >>= 1) {
    if (tid < off) red[tid] = fmaxf(red[tid], red[tid + off]);
    __syncthreads();
  }
  float mx = red[0]; __syncthreads();
  float lsum = 0.f;
  for (int i = tid; i < Vv; i += 1024) lsum += expf(row[i] - mx);
  red[tid] = lsum; __syncthreads();
  for (int off = 512; off; off >>= 1) {
    if (tid < off) red[tid] += red[tid + off];
    __syncthreads();
  }
  float lse = mx + logf(red[0]);
  for (int i = tid; i < Vv; i += 1024) row[i] -= lse;
}

// ---------------------------- host ----------------------------
extern "C" void kernel_launch(void* const* d_in, const int* in_sizes, int n_in,
                              void* d_out, int out_size, void* d_ws, size_t ws_size,
                              hipStream_t stream) {
  const float* enc_out = (const float*)d_in[0];
  const float* enc_h   = (const float*)d_in[1];
  const float* enc_c   = (const float*)d_in[2];
  const int*   ques    = (const int*)d_in[3];
  const float* emb     = (const float*)d_in[4];
  const float* Wih     = (const float*)d_in[5];
  const float* Whh     = (const float*)d_in[6];
  const float* bih     = (const float*)d_in[7];
  const float* bhh     = (const float*)d_in[8];
  const float* Wai     = (const float*)d_in[9];
  const float* Wao     = (const float*)d_in[10];
  const float* Wgen    = (const float*)d_in[11];
  const float* bgen    = (const float*)d_in[12];
  float* out = (float*)d_out;

  char* base = (char*)d_ws;
  size_t off = 0;
  auto carve = [&](size_t bytes) {
    char* p = base + off;
    off = (off + bytes + 255) & ~(size_t)255;
    return p;
  };
  bf16*  wcat  = (bf16*)carve((size_t)NG * KX * 2);
  float* biasc = (float*)carve((size_t)NG * 4);
  bf16*  wai   = (bf16*)carve((size_t)KH * KH * 2);
  bf16*  wao   = (bf16*)carve((size_t)KH * KC * 2);
  bf16*  xcat  = (bf16*)carve((size_t)Bb * KX * 2);
  float* gates = (float*)carve((size_t)Bb * NG * 4);
  float* h     = (float*)carve((size_t)Bb * Hh * 4);
  float* c     = (float*)carve((size_t)Bb * Hh * 4);
  float* feed  = (float*)carve((size_t)Bb * Hh * 4);
  bf16*  h_bf  = (bf16*)carve((size_t)Bb * KH * 2);
  float* q     = (float*)carve((size_t)Bb * KH * 4);
  float* align = (float*)carve((size_t)Bb * Ss * 4);
  bf16*  ctxh  = (bf16*)carve((size_t)Bb * KC * 2);
  bf16*  attnh = (bf16*)carve((size_t)Bb * KH * 2);
  bf16*  wgenb = (bf16*)carve((size_t)Vv * KH * 2);  // 60.8 MB, optional
  bool use_wgenb = (off <= ws_size);                 // ws_size fixed -> deterministic
  (void)in_sizes; (void)n_in; (void)out_size;

  // one-time prep (re-run every call; inputs unchanged -> deterministic)
  k_prep_wcat<<<(NG * KX + 255) / 256, 256, 0, stream>>>(Wih, Whh, bih, bhh, wcat, biasc);
  k_prep_wai<<<(KH * KH + 255) / 256, 256, 0, stream>>>(Wai, wai);
  k_prep_wao<<<(KH * KC + 255) / 256, 256, 0, stream>>>(Wao, wao);
  k_init_state<<<(Bb * Hh + 255) / 256, 256, 0, stream>>>(enc_h, enc_c, h, c, feed);
  if (use_wgenb) {
    long n = (long)Vv * KH;
    k_prep_wgen<<<(unsigned)((n + 255) / 256), 256, 0, stream>>>(Wgen, wgenb);
  }

  for (int t = 0; t < Tt - 1; ++t) {
    k_build_xcat<<<(Bb * KX + 255) / 256, 256, 0, stream>>>(ques, emb, feed, h, xcat, t);
    k_gemm_gates<<<NG / 80, 32, 0, stream>>>(xcat, wcat, biasc, gates);
    k_lstm_ew<<<(Bb * 624 + 255) / 256, 256, 0, stream>>>(gates, h, c, h_bf, ctxh);
    k_gemm_q<<<KH / 32, 32, 0, stream>>>(h_bf, wai, q);
    k_attn_score<<<(Bb * Ss) / 8, 256, 0, stream>>>(q, enc_out, align);
    k_softmax_ctx<<<Bb, 512, 0, stream>>>(align, enc_out, ctxh);
    k_gemm_attnout<<<KH / 32, 32, 0, stream>>>(ctxh, wao, feed, attnh);
    if (use_wgenb)
      k_gemm_logits_bf<<<Vv / 80, 32, 0, stream>>>(attnh, wgenb, bgen, out, t);
    else
      k_gemm_logits_f32<<<dim3(Vv / 80, 2), 32, 0, stream>>>(attnh, Wgen, bgen, out, t);
    k_logsoftmax<<<Bb, 1024, 0, stream>>>(out, t);
  }
}
